// Net_66185446031452
// MI455X (gfx1250) — matrix operand order repaired and checked by reading, not compiled, not run on hardware
//
#include <hip/hip_runtime.h>
#include <hip/hip_bf16.h>
#include <math.h>

#define BB   128
#define NP_  256
#define NN   (BB * NP_)        // 32768 nodes
#define EPD  2048
#define EE   (2 * BB * EPD)    // 524288 symmetric edges
#define FH   128               // F_IN == H == 128
#define DIMS 128
#define C1_  32
#define C2_  64
#define NC_  10
#define HEADS_ 8
#define DK_  16

typedef __attribute__((ext_vector_type(16))) _Float16 v16h;
typedef __attribute__((ext_vector_type(8)))  _Float16 v8h;
typedef __attribute__((ext_vector_type(8)))  float    v8f;
typedef __attribute__((__vector_size__(4 * sizeof(int)))) int v4i_;

#if defined(__has_builtin)
#if __has_builtin(__builtin_amdgcn_global_load_async_to_lds_b128) && \
    __has_builtin(__builtin_amdgcn_s_wait_asynccnt)
#define USE_ASYNC_LDS 1
#endif
#endif
#ifndef USE_ASYNC_LDS
#define USE_ASYNC_LDS 0
#endif

// ---------------- elementwise / setup kernels ----------------

__global__ void k_zero(float* __restrict__ p, long n) {
  long i = (long)blockIdx.x * blockDim.x + threadIdx.x;
  if (i < n) p[i] = 0.0f;
}

__global__ void k_deg(const int* __restrict__ dst, float* __restrict__ deg) {
  int i = blockIdx.x * blockDim.x + threadIdx.x;
  if (i < EE) atomicAdd(&deg[dst[i]], 1.0f);
}

__global__ void k_dnorm(float* __restrict__ deg) {
  int i = blockIdx.x * blockDim.x + threadIdx.x;
  if (i < NN) deg[i] = rsqrtf(fmaxf(deg[i], 1.0f));   // in-place: deg -> dnorm
}

// o = (f32 -> f16)(h * dnorm[row])
__global__ void k_scaled_f16(const float* __restrict__ h, const float* __restrict__ dnorm,
                             _Float16* __restrict__ o, long n) {
  long i = (long)blockIdx.x * blockDim.x + threadIdx.x;
  if (i < n) o[i] = (_Float16)(h[i] * dnorm[i >> 7]);  // FH == 128
}

__global__ void k_f16(const float* __restrict__ a, _Float16* __restrict__ o, long n) {
  long i = (long)blockIdx.x * blockDim.x + threadIdx.x;
  if (i < n) o[i] = (_Float16)a[i];
}

// ---------------- weight pack: row-major f32 K(128) x N(128) -> WMMA B-fragment layout ----
// Bp[((kc*8 + tn)*32 + lane)*16 + i] = W[(kc*32 + 16*(lane>>4) + i) * 128 + tn*16 + (lane&15)]
__global__ void k_packB(const float* __restrict__ W, _Float16* __restrict__ Bp) {
  int t = blockIdx.x * blockDim.x + threadIdx.x;      // 0 .. 16383
  int i    = t & 15;
  int lane = (t >> 4) & 31;
  int tn   = (t >> 9) & 7;
  int kc   = t >> 12;
  int k = kc * 32 + ((lane >> 4) << 4) + i;
  int n = tn * 16 + (lane & 15);
  Bp[t] = (_Float16)W[k * FH + n];
}

// ---------------- WMMA GEMM: C[M,128] = A[M,128] @ B[128,128], branchless epilogue --------
// Block: 256 threads = 8 waves. Whole packed weight (32KB) staged in LDS (async path on
// CDNA5: global_load_async_to_lds_b128 + s_wait_asynccnt; A rows prefetched meanwhile).
// Each wave computes one 16-row strip across all 8 n-tiles: 8 v8f accumulators,
// 4 K-chunks x 8 n-tiles = 32 v_wmma per wave. A fragment = two b128 loads per chunk;
// B fragment = two ds_load_b128 from the packed LDS image.

__global__ void __launch_bounds__(256)
k_gemm128(const _Float16* __restrict__ A, const _Float16* __restrict__ Bp,
          const float* __restrict__ bias, float* __restrict__ C,
          int M, float lo) {
  __shared__ _Float16 Bs[4 * 8 * 32 * 16];            // 16384 halves = 32 KB

  int tid  = threadIdx.x;
  int lane = tid & 31;
  int wave = tid >> 5;

  int tm    = blockIdx.x * 8 + wave;                  // m-tile (16 rows)
  int hi    = lane >> 4;
  int ln15  = lane & 15;
  int row_a = tm * 16 + ln15;

  // get this lane's A row in flight while B is staged into LDS
  __builtin_prefetch(A + (long)row_a * FH, 0, 3);

  // stage packed B into LDS (2048 x 16B)
#if USE_ASYNC_LDS
  {
    typedef __attribute__((address_space(1))) v4i_ as1_v4i;
    typedef __attribute__((address_space(3))) v4i_ as3_v4i;
    as1_v4i* g = (as1_v4i*)Bp;       // explicit addrspacecast (drops const)
    as3_v4i* s = (as3_v4i*)&Bs[0];
    for (int t = tid; t < 2048; t += 256)
      __builtin_amdgcn_global_load_async_to_lds_b128(g + t, s + t, 0, 0);
    __builtin_amdgcn_s_wait_asynccnt(0);
  }
#else
  {
    const uint4* g = (const uint4*)Bp;
    uint4*       s = (uint4*)Bs;
    for (int t = tid; t < 2048; t += 256) s[t] = g[t];
  }
#endif
  __syncthreads();

  if (row_a >= M) return;

  const v8h* Ar = (const v8h*)(A + (long)row_a * FH); // 16 v8h per row

  v8f acc[8];
#pragma unroll
  for (int tn = 0; tn < 8; ++tn) acc[tn] = (v8f){};

#pragma unroll
  for (int kc = 0; kc < 4; ++kc) {
    v8h a0 = Ar[kc * 4 + 2 * hi];
    v8h a1 = Ar[kc * 4 + 2 * hi + 1];
    v16h a = __builtin_shufflevector(a0, a1, 0, 1, 2, 3, 4, 5, 6, 7,
                                     8, 9, 10, 11, 12, 13, 14, 15);
#pragma unroll
    for (int tn = 0; tn < 8; ++tn) {
      const v8h* bl = (const v8h*)&Bs[(((kc * 8 + tn) * 32) + lane) * 16];
      v8h b0 = bl[0];
      v8h b1 = bl[1];
      v16h b = __builtin_shufflevector(b0, b1, 0, 1, 2, 3, 4, 5, 6, 7,
                                       8, 9, 10, 11, 12, 13, 14, 15);
      acc[tn] = __builtin_amdgcn_wmma_f32_16x16x32_f16(false, a, false, b,
                                                       (short)0, acc[tn], false, false);
    }
  }

#pragma unroll
  for (int tn = 0; tn < 8; ++tn) {
    int col = tn * 16 + ln15;
    float bv = bias[col];
#pragma unroll
    for (int r = 0; r < 8; ++r) {
      int row = tm * 16 + r + (hi << 3);
      C[(long)row * FH + col] = fmaxf(acc[tn][r] + bv, lo);
    }
  }
}

// ---------------- GraphConv aggregation (float4 gather + 4 atomics/thread) ----------------

__global__ void k_edge_agg(const int* __restrict__ src, const int* __restrict__ dst,
                           const float* __restrict__ hw, float* __restrict__ agg) {
  long i = (long)blockIdx.x * blockDim.x + threadIdx.x;
  if (i < (long)EE * (FH / 4)) {
    int e  = (int)(i >> 5);
    int f4 = (int)(i & 31) << 2;
    float4 hv = *(const float4*)&hw[(long)src[e] * FH + f4];
    float* d  = &agg[(long)dst[e] * FH + f4];
    atomicAdd(d + 0, hv.x);
    atomicAdd(d + 1, hv.y);
    atomicAdd(d + 2, hv.z);
    atomicAdd(d + 3, hv.w);
  }
}

__global__ void k_gconv_fin(const float* __restrict__ agg, const float* __restrict__ dnorm,
                            const float* __restrict__ bias, float* __restrict__ h) {
  long i = (long)blockIdx.x * blockDim.x + threadIdx.x;
  if (i < (long)NN * FH)
    h[i] = fmaxf(agg[i] * dnorm[i >> 7] + bias[i & 127], 0.0f);
}

// ---------------- SAGPool score path ----------------

__global__ void k_score_pre(const float* __restrict__ h, const float* __restrict__ dnorm,
                            const float* __restrict__ Wp, float* __restrict__ s1) {
  int n = blockIdx.x * blockDim.x + threadIdx.x;
  if (n < NN) {
    const float4* hp = (const float4*)&h[(long)n * FH];
    const float4* wp = (const float4*)Wp;
    float acc = 0.0f;
#pragma unroll 8
    for (int f = 0; f < FH / 4; ++f) {
      float4 hv = hp[f], wv = wp[f];
      acc += hv.x * wv.x + hv.y * wv.y + hv.z * wv.z + hv.w * wv.w;
    }
    s1[n] = acc * dnorm[n];
  }
}

__global__ void k_score_edge(const int* __restrict__ src, const int* __restrict__ dst,
                             const float* __restrict__ s1, float* __restrict__ s2) {
  int i = blockIdx.x * blockDim.x + threadIdx.x;
  if (i < EE) atomicAdd(&s2[dst[i]], s1[src[i]]);
}

__global__ void k_score_fin(const float* __restrict__ s2, const float* __restrict__ dnorm,
                            const float* __restrict__ bp, float* __restrict__ score) {
  int i = blockIdx.x * blockDim.x + threadIdx.x;
  if (i < NN) score[i] = s2[i] * dnorm[i] + bp[0];
}

// exact stable top-k by rank (matches jax.lax.top_k ordering): one block per graph
__global__ void k_topk(const float* __restrict__ score, float* __restrict__ vals,
                       int* __restrict__ idxs) {
  int b = blockIdx.x, i = threadIdx.x;
  __shared__ float s[NP_];
  s[i] = score[b * NP_ + i];
  __syncthreads();
  float si = s[i];
  int rank = 0;
  for (int j = 0; j < NP_; ++j) {
    float sj = s[j];
    rank += (sj > si) || (sj == si && j < i);
  }
  if (rank < DIMS) { vals[b * DIMS + rank] = si; idxs[b * DIMS + rank] = i; }
}

__global__ void k_pool(const float* __restrict__ h, const float* __restrict__ vals,
                       const int* __restrict__ idxs, float* __restrict__ xp) {
  long t = (long)blockIdx.x * blockDim.x + threadIdx.x;
  if (t < (long)BB * DIMS * FH) {
    int f = (int)(t & 127);
    int d = (int)((t >> 7) & 127);
    int b = (int)(t >> 14);
    int node = b * NP_ + idxs[b * DIMS + d];
    xp[t] = h[(long)node * FH + f] * tanhf(vals[b * DIMS + d]);
  }
}

// ---------------- MHSA core: grid (B, HEADS), 128 threads (one per query row) ----------------

__global__ void k_attn(const float* __restrict__ q, const float* __restrict__ k,
                       const float* __restrict__ v, float* __restrict__ o) {
  int b = blockIdx.x, hh = blockIdx.y, i = threadIdx.x;
  __shared__ float Ks[DIMS][DK_];
  __shared__ float Vs[DIMS][DK_];
  long base = (long)b * DIMS * FH + hh * DK_;
#pragma unroll
  for (int d = 0; d < DK_; ++d) {
    Ks[i][d] = k[base + (long)i * FH + d];
    Vs[i][d] = v[base + (long)i * FH + d];
  }
  __syncthreads();
  float qi[DK_];
#pragma unroll
  for (int d = 0; d < DK_; ++d) qi[d] = q[base + (long)i * FH + d];
  const float scale = 0.25f;  // 1/sqrt(16)
  float mx = -1e30f;
  for (int m = 0; m < DIMS; ++m) {
    float s = 0.0f;
#pragma unroll
    for (int d = 0; d < DK_; ++d) s += qi[d] * Ks[m][d];
    mx = fmaxf(mx, s * scale);
  }
  float sum = 0.0f, acc[DK_];
#pragma unroll
  for (int d = 0; d < DK_; ++d) acc[d] = 0.0f;
  for (int m = 0; m < DIMS; ++m) {
    float s = 0.0f;
#pragma unroll
    for (int d = 0; d < DK_; ++d) s += qi[d] * Ks[m][d];
    float w = expf(s * scale - mx);
    sum += w;
#pragma unroll
    for (int d = 0; d < DK_; ++d) acc[d] += w * Vs[m][d];
  }
  float inv = 1.0f / sum;
#pragma unroll
  for (int d = 0; d < DK_; ++d) o[base + (long)i * FH + d] = acc[d] * inv;
}

// ---------------- classifier tail ----------------

__global__ void k_conv1(const float* __restrict__ feats, const float* __restrict__ Wc1,
                        const float* __restrict__ bc1, float* __restrict__ y) {
  int b = blockIdx.x, o = blockIdx.y, w = threadIdx.x;
  float acc = bc1[o];
  for (int c = 0; c < 3; ++c)
    for (int d = 0; d < DIMS; ++d)
      acc += feats[(((long)c * BB + b) * DIMS + d) * FH + w] * Wc1[(o * 3 + c) * DIMS + d];
  y[((long)b * C1_ + o) * FH + w] = fmaxf(acc, 0.0f);
}

__global__ void k_conv2(const float* __restrict__ y, const float* __restrict__ Wc2,
                        const float* __restrict__ bc2, float* __restrict__ z) {
  int b = blockIdx.x, o = threadIdx.x;
  float acc = bc2[o];
  for (int c = 0; c < C1_; ++c)
    for (int w = 0; w < FH / 2; ++w) {
      const float* row = &y[((long)b * C1_ + c) * FH + 2 * w];
      acc += fmaxf(row[0], row[1]) * Wc2[(o * C1_ + c) * (FH / 2) + w];
    }
  z[b * C2_ + o] = fmaxf(acc, 0.0f);
}

__global__ void k_head(const float* __restrict__ z, const float* __restrict__ Wl1,
                       const float* __restrict__ bl1, const float* __restrict__ Wl2,
                       const float* __restrict__ bl2, float* __restrict__ out) {
  int b = blockIdx.x, j = threadIdx.x;  // 64 threads
  __shared__ float zs[C2_], t1[C2_], t2[NC_];
  zs[j] = z[b * C2_ + j];
  __syncthreads();
  float acc = bl1[j];
  for (int i = 0; i < C2_; ++i) acc += zs[i] * Wl1[i * C2_ + j];
  t1[j] = fmaxf(acc, 0.0f);
  __syncthreads();
  if (j < NC_) {
    float a = bl2[j];
    for (int i = 0; i < C2_; ++i) a += t1[i] * Wl2[i * NC_ + j];
    t2[j] = a;
  }
  __syncthreads();
  if (j == 0) {
    float mx = t2[0];
    for (int jj = 1; jj < NC_; ++jj) mx = fmaxf(mx, t2[jj]);
    float se = 0.0f;
    for (int jj = 0; jj < NC_; ++jj) se += expf(t2[jj] - mx);
    float l = mx + logf(se);
    for (int jj = 0; jj < NC_; ++jj) out[b * NC_ + jj] = t2[jj] - l;
  }
}

// ---------------- host side ----------------

static inline void* bump(char*& p, size_t bytes) {
  void* r = (void*)p;
  p += (bytes + 255) & ~(size_t)255;
  return r;
}

extern "C" void kernel_launch(void* const* d_in, const int* in_sizes, int n_in,
                              void* d_out, int out_size, void* d_ws, size_t ws_size,
                              hipStream_t stream) {
  const float* x    = (const float*)d_in[0];
  const int*   src  = (const int*)d_in[1];
  const int*   dst  = (const int*)d_in[2];
  const float* W_g[3] = {(const float*)d_in[3], (const float*)d_in[5], (const float*)d_in[7]};
  const float* b_g[3] = {(const float*)d_in[4], (const float*)d_in[6], (const float*)d_in[8]};
  const float* W_p  = (const float*)d_in[9];
  const float* b_p  = (const float*)d_in[10];
  const float* W_q  = (const float*)d_in[11];
  const float* W_k  = (const float*)d_in[12];
  const float* W_v  = (const float*)d_in[13];
  const float* W_c1 = (const float*)d_in[14];
  const float* b_c1 = (const float*)d_in[15];
  const float* W_c2 = (const float*)d_in[16];
  const float* b_c2 = (const float*)d_in[17];
  const float* W_l1 = (const float*)d_in[18];
  const float* b_l1 = (const float*)d_in[19];
  const float* W_l2 = (const float*)d_in[20];
  const float* b_l2 = (const float*)d_in[21];
  float* out = (float*)d_out;

  char* p = (char*)d_ws;
  float*    dnorm = (float*)bump(p, (size_t)NN * 4);
  float*    h     = (float*)bump(p, (size_t)NN * FH * 4);
  float*    agg   = (float*)bump(p, (size_t)NN * FH * 4);   // reused as K proj
  float*    hw    = (float*)bump(p, (size_t)NN * FH * 4);   // reused as Q proj
  _Float16* a16   = (_Float16*)bump(p, (size_t)NN * FH * 2);
  _Float16* Bp_g  = (_Float16*)bump(p, (size_t)FH * FH * 2);
  _Float16* Bp_q  = (_Float16*)bump(p, (size_t)FH * FH * 2);
  _Float16* Bp_k  = (_Float16*)bump(p, (size_t)FH * FH * 2);
  _Float16* Bp_v  = (_Float16*)bump(p, (size_t)FH * FH * 2);
  float*    zbias = (float*)bump(p, (size_t)FH * 4);
  float*    s1    = (float*)bump(p, (size_t)NN * 4);
  float*    s2    = (float*)bump(p, (size_t)NN * 4);
  float*    score = (float*)bump(p, (size_t)NN * 4);
  float*    vals  = (float*)bump(p, (size_t)BB * DIMS * 4);
  int*      idxs  = (int*)bump(p, (size_t)BB * DIMS * 4);
  float*    xp    = (float*)bump(p, (size_t)BB * DIMS * FH * 4);
  _Float16* xp16  = (_Float16*)bump(p, (size_t)BB * DIMS * FH * 2);
  float*    vproj = (float*)bump(p, (size_t)BB * DIMS * FH * 4);
  float*    feats = (float*)bump(p, (size_t)3 * BB * DIMS * FH * 4);
  float*    ybuf  = (float*)bump(p, (size_t)BB * C1_ * FH * 4);
  float*    zbuf  = (float*)bump(p, (size_t)BB * C2_ * 4);

  const long NH  = (long)NN * FH;        // 4,194,304
  const long PH  = (long)BB * DIMS * FH; // 2,097,152
  const long E4  = (long)EE * (FH / 4);  // 16,777,216 (edge x float4)
  const float NEG = -__builtin_inff();

  // degree -> dnorm, zero bias for no-bias GEMMs
  k_zero<<<(NN + 255) / 256, 256, 0, stream>>>(dnorm, NN);
  k_zero<<<1, 256, 0, stream>>>(zbias, FH);
  k_deg<<<EE / 256, 256, 0, stream>>>(dst, dnorm);
  k_dnorm<<<NN / 256, 256, 0, stream>>>(dnorm);

  // pack attention weights once
  k_packB<<<FH * FH / 256, 256, 0, stream>>>(W_q, Bp_q);
  k_packB<<<FH * FH / 256, 256, 0, stream>>>(W_k, Bp_k);
  k_packB<<<FH * FH / 256, 256, 0, stream>>>(W_v, Bp_v);

  const float* hcur = x;
  for (int L = 0; L < 3; ++L) {
    // GraphConv: h = relu(dnorm * segsum((hcur*dnorm)@W) + b)
    k_scaled_f16<<<(int)((NH + 255) / 256), 256, 0, stream>>>(hcur, dnorm, a16, NH);
    k_packB<<<FH * FH / 256, 256, 0, stream>>>(W_g[L], Bp_g);
    k_gemm128<<<NN / 128, 256, 0, stream>>>(a16, Bp_g, zbias, hw, NN, NEG);
    k_zero<<<(int)((NH + 255) / 256), 256, 0, stream>>>(agg, NH);
    k_edge_agg<<<(int)(E4 / 256), 256, 0, stream>>>(src, dst, hw, agg);
    k_gconv_fin<<<(int)((NH + 255) / 256), 256, 0, stream>>>(agg, dnorm, b_g[L], h);

    // SAGPool score = gconv(h, W_p, b_p)
    k_score_pre<<<NN / 256, 256, 0, stream>>>(h, dnorm, W_p, s1);
    k_zero<<<(NN + 255) / 256, 256, 0, stream>>>(s2, NN);
    k_score_edge<<<EE / 256, 256, 0, stream>>>(src, dst, s1, s2);
    k_score_fin<<<NN / 256, 256, 0, stream>>>(s2, dnorm, b_p, score);
    k_topk<<<BB, NP_, 0, stream>>>(score, vals, idxs);
    k_pool<<<(int)((PH + 255) / 256), 256, 0, stream>>>(h, vals, idxs, xp);

    // MHSA: Q/K/V projections via WMMA, attention core in VALU
    k_f16<<<(int)((PH + 255) / 256), 256, 0, stream>>>(xp, xp16, PH);
    k_gemm128<<<BB * DIMS / 128, 256, 0, stream>>>(xp16, Bp_q, zbias, hw, BB * DIMS, NEG);
    k_gemm128<<<BB * DIMS / 128, 256, 0, stream>>>(xp16, Bp_k, zbias, agg, BB * DIMS, NEG);
    k_gemm128<<<BB * DIMS / 128, 256, 0, stream>>>(xp16, Bp_v, zbias, vproj, BB * DIMS, NEG);
    k_attn<<<dim3(BB, HEADS_), DIMS, 0, stream>>>(hw, agg, vproj, feats + (long)L * PH);

    hcur = h;
  }

  // classifier
  k_conv1<<<dim3(BB, C1_), FH, 0, stream>>>(feats, W_c1, b_c1, ybuf);
  k_conv2<<<BB, C2_, 0, stream>>>(ybuf, W_c2, b_c2, zbuf);
  k_head<<<BB, C2_, 0, stream>>>(zbuf, W_l1, b_l1, W_l2, b_l2, out);
}
